// MSN_72112500900176
// MI455X (gfx1250) — matrix operand order repaired and checked by reading, not compiled, not run on hardware
//
#include <hip/hip_runtime.h>
#include <hip/hip_bf16.h>

#define EPSV   1e-5f
#define RNUM   16
#define NPTS   2048
#define NCAB   8
#define BATCH  2
#define CHOUT  1043   // 4*256 + 19 output channels

typedef __attribute__((ext_vector_type(16))) _Float16 v16h;
typedef __attribute__((ext_vector_type(8)))  float    v8f;

// ---------------------------------------------------------------------------
// Layer 1: 3 -> 64 pointwise conv + BN + ReLU (K too small for WMMA)
// ---------------------------------------------------------------------------
__global__ __launch_bounds__(256)
void layer1_kernel(const float* __restrict__ x, const float* __restrict__ w,
                   const float* __restrict__ bias,
                   const float* __restrict__ g, const float* __restrict__ be,
                   const float* __restrict__ m, const float* __restrict__ v,
                   float* __restrict__ h1)
{
    int id = blockIdx.x * blockDim.x + threadIdx.x;   // b*64*N + o*N + n
    int n = id & (NPTS - 1);
    int o = (id >> 11) & 63;
    int b = id >> 17;
    if (b >= BATCH) return;
    const float* xb = x + (size_t)b * 3 * NPTS;
    float acc = bias[o];
    #pragma unroll
    for (int c = 0; c < 3; ++c) acc += w[o * 3 + c] * xb[(size_t)c * NPTS + n];
    float s = g[o] * rsqrtf(v[o] + EPSV);
    acc = (acc - m[o]) * s + be[o];
    h1[((size_t)b * 64 + o) * NPTS + n] = fmaxf(acc, 0.f);
}

// ---------------------------------------------------------------------------
// WMMA pointwise conv: out[o,n] = act(BN(W @ in + bias))
// one wave (32 threads) per 16x16 output tile; K accumulated 32 at a time
// via v_wmma_f32_16x16x32_f16.  A/B use contiguous k = 16*(lane/16) + e.
// ---------------------------------------------------------------------------
template<int CIN, int COUT, bool DO_BN, bool DO_RELU>
__global__ __launch_bounds__(32)
void wmma_pointwise_conv(const float* __restrict__ in,   // [B, CIN, N]
                         const float* __restrict__ w,    // [COUT, CIN]
                         const float* __restrict__ bias, // [COUT]
                         const float* __restrict__ bg, const float* __restrict__ bb,
                         const float* __restrict__ bm, const float* __restrict__ bv,
                         float* __restrict__ out, int N)
{
    const int lane = threadIdx.x;        // 0..31
    const int hi   = lane >> 4;
    const int lr   = lane & 15;
    const int n0   = blockIdx.x * 16;
    const int o0   = blockIdx.y * 16;
    const int b    = blockIdx.z;

    const float* inb  = in  + (size_t)b * CIN  * N;
    float*       outb = out + (size_t)b * COUT * N;

    v8f acc = {};
    for (int c0 = 0; c0 < CIN; c0 += 32) {
        v16h a, bt;
        const float* wrow = w + (size_t)(o0 + lr) * CIN + c0;
        #pragma unroll
        for (int e = 0; e < 16; ++e) {
            const int vv = e >> 1, j = e & 1;
            // A (16x32, f16): M = lane%16; K = 2v+j (+16 for hi lanes)
            const int ka = 2 * vv + j + (hi << 4);
            a[e] = (_Float16)wrow[ka];
            // B (32x16, f16): N = lane%16; K = 2v+j (+16 for hi lanes)
            const int kb = 2 * vv + j + (hi << 4);
            bt[e] = (_Float16)inb[(size_t)(c0 + kb) * N + n0 + lr];
        }
        acc = __builtin_amdgcn_wmma_f32_16x16x32_f16(false, a, false, bt,
                                                     (short)0, acc, false, false);
    }
    #pragma unroll
    for (int r2 = 0; r2 < 8; ++r2) {
        const int o = o0 + r2 + hi * 8;          // D: VGPR r holds M=r / M=r+8
        float val = acc[r2] + bias[o];
        if (DO_BN) {
            float s = bg[o] * rsqrtf(bv[o] + EPSV);
            val = (val - bm[o]) * s + bb[o];
        }
        if (DO_RELU) val = fmaxf(val, 0.f);
        outb[(size_t)o * N + n0 + lr] = val;
    }
}

// ---------------------------------------------------------------------------
// id_activa = argmax over R (first max wins, matching jnp.argmax)
// ---------------------------------------------------------------------------
__global__ __launch_bounds__(256)
void argmax_kernel(const float* __restrict__ val, int* __restrict__ ida)
{
    int id = blockIdx.x * blockDim.x + threadIdx.x;   // b*N + n
    int n = id & (NPTS - 1);
    int b = id >> 11;
    if (b >= BATCH) return;
    const float* vb = val + (size_t)b * RNUM * NPTS + n;
    float best = vb[0];
    int bi = 0;
    #pragma unroll
    for (int r = 1; r < RNUM; ++r) {
        float t = vb[(size_t)r * NPTS];
        if (t > best) { best = t; bi = r; }
    }
    ida[id] = bi;
}

// ---------------------------------------------------------------------------
// Per-(b,r) stable argsort, descending by val (tie -> smaller index first).
// Bitonic sort in LDS; the composite comparator is a strict total order, so
// the result equals jnp.argsort(-val, kind='stable').
// ---------------------------------------------------------------------------
__global__ __launch_bounds__(1024)
void sort_kernel(const float* __restrict__ val, int* __restrict__ idx)
{
    __shared__ float kv[NPTS];
    __shared__ int   ki[NPTS];
    const int r = blockIdx.x, b = blockIdx.y;
    const float* row = val + ((size_t)b * RNUM + r) * NPTS;
    const int t = threadIdx.x;
    kv[t] = row[t];               ki[t] = t;
    kv[t + 1024] = row[t + 1024]; ki[t + 1024] = t + 1024;
    __syncthreads();
    for (int k = 2; k <= NPTS; k <<= 1) {
        for (int j = k >> 1; j > 0; j >>= 1) {
            #pragma unroll
            for (int half = 0; half < 2; ++half) {
                const int i = t + half * 1024;
                const int ixj = i ^ j;
                if (ixj > i) {
                    const bool up = ((i & k) == 0);
                    float va = kv[i], vb2 = kv[ixj];
                    int   ia = ki[i], ib  = ki[ixj];
                    // "a after b" in desired order (val desc, idx asc on tie)
                    bool a_after_b = (va < vb2) || (va == vb2 && ia > ib);
                    if (a_after_b == up) {
                        kv[i] = vb2; kv[ixj] = va;
                        ki[i] = ib;  ki[ixj] = ia;
                    }
                }
            }
            __syncthreads();
        }
    }
    int* orow = idx + ((size_t)b * RNUM + r) * NPTS;
    orow[t] = ki[t];
    orow[t + 1024] = ki[t + 1024];
}

// ---------------------------------------------------------------------------
// Fused gather: sp_cube (ch 0..255), cabins (chunk max), sp_windows (256..511)
// One block per (c, r, b); idx row staged in LDS; b128 stores on the output.
// ---------------------------------------------------------------------------
__global__ __launch_bounds__(256)
void gather_kernel(const float* __restrict__ h3, const int* __restrict__ idx,
                   float* __restrict__ cabins, float* __restrict__ out)
{
    __shared__ int   sidx[NPTS];
    __shared__ float smax[256];
    __shared__ float scab[NCAB];
    const int c = blockIdx.x, r = blockIdx.y, b = blockIdx.z;
    const int t = threadIdx.x;
    const int4* irow4 = (const int4*)(idx + ((size_t)b * RNUM + r) * NPTS);
    int4* sidx4 = (int4*)sidx;
    sidx4[t * 2]     = irow4[t * 2];
    sidx4[t * 2 + 1] = irow4[t * 2 + 1];
    __syncthreads();
    const float* hrow = h3 + ((size_t)b * 256 + c) * NPTS;
    float* orow = out + (((size_t)b * CHOUT + c) * RNUM + r) * NPTS;   // sp_cube
    float lmax = -3.4e38f;
    float vv[8];
    #pragma unroll
    for (int u = 0; u < 8; ++u) {
        const int n = t * 8 + u;             // chunk = t/32 (constant over u)
        vv[u] = hrow[sidx[n]];
        lmax = fmaxf(lmax, vv[u]);
    }
    {
        float4* orow4 = (float4*)orow;
        float4 p0 = { vv[0], vv[1], vv[2], vv[3] };
        float4 p1 = { vv[4], vv[5], vv[6], vv[7] };
        orow4[t * 2]     = p0;
        orow4[t * 2 + 1] = p1;
    }
    smax[t] = lmax;
    __syncthreads();
    if (t < NCAB) {
        float mx = smax[t * 32];
        for (int q = 1; q < 32; ++q) mx = fmaxf(mx, smax[t * 32 + q]);
        scab[t] = mx;
        cabins[(((size_t)b * 256 + c) * RNUM + r) * NCAB + t] = mx;
    }
    __syncthreads();
    float4* wrow4 = (float4*)(out + (((size_t)b * CHOUT + 256 + c) * RNUM + r) * NPTS);
    const float cabv = scab[t >> 5];
    const float4 cb = { cabv, cabv, cabv, cabv };
    wrow4[t * 2]     = cb;
    wrow4[t * 2 + 1] = cb;
}

// ---------------------------------------------------------------------------
// trains: three convs along the 8-cabin axis (1x3, 1x3, 1x4), 256ch each.
// One block per (r, b); thread = output channel.
// ---------------------------------------------------------------------------
__global__ __launch_bounds__(256)
void trains_kernel(const float* __restrict__ cabins,
                   const float* __restrict__ w1, const float* __restrict__ b1,
                   const float* __restrict__ w2, const float* __restrict__ b2,
                   const float* __restrict__ w3, const float* __restrict__ b3,
                   float* __restrict__ trains)
{
    __shared__ float cab[256][NCAB];
    __shared__ float t1[256][6];
    __shared__ float t2[256][4];
    const int r = blockIdx.x, b = blockIdx.y;
    const int o = threadIdx.x;
    const float* crow = cabins + (((size_t)b * 256 + o) * RNUM + r) * NCAB;
    #pragma unroll
    for (int k = 0; k < NCAB; ++k) cab[o][k] = crow[k];
    __syncthreads();
    float a1[6];
    #pragma unroll
    for (int p = 0; p < 6; ++p) a1[p] = b1[o];
    for (int i = 0; i < 256; ++i) {
        const float w0 = w1[(o * 256 + i) * 3 + 0];
        const float wA = w1[(o * 256 + i) * 3 + 1];
        const float wB = w1[(o * 256 + i) * 3 + 2];
        #pragma unroll
        for (int p = 0; p < 6; ++p)
            a1[p] += w0 * cab[i][p] + wA * cab[i][p + 1] + wB * cab[i][p + 2];
    }
    #pragma unroll
    for (int p = 0; p < 6; ++p) t1[o][p] = a1[p];
    __syncthreads();
    float a2[4];
    #pragma unroll
    for (int p = 0; p < 4; ++p) a2[p] = b2[o];
    for (int i = 0; i < 256; ++i) {
        const float w0 = w2[(o * 256 + i) * 3 + 0];
        const float wA = w2[(o * 256 + i) * 3 + 1];
        const float wB = w2[(o * 256 + i) * 3 + 2];
        #pragma unroll
        for (int p = 0; p < 4; ++p)
            a2[p] += w0 * t1[i][p] + wA * t1[i][p + 1] + wB * t1[i][p + 2];
    }
    #pragma unroll
    for (int p = 0; p < 4; ++p) t2[o][p] = a2[p];
    __syncthreads();
    float a3 = b3[o];
    for (int i = 0; i < 256; ++i) {
        #pragma unroll
        for (int k = 0; k < 4; ++k) a3 += w3[(o * 256 + i) * 4 + k] * t2[i][k];
    }
    trains[((size_t)b * 256 + o) * RNUM + r] = a3;
}

// ---------------------------------------------------------------------------
// station: 16x1 conv collapsing the region axis of trains
// ---------------------------------------------------------------------------
__global__ __launch_bounds__(256)
void station_kernel(const float* __restrict__ trains, const float* __restrict__ w5,
                    const float* __restrict__ b5, float* __restrict__ station)
{
    __shared__ float tr[256][RNUM];
    const int b = blockIdx.x, o = threadIdx.x;
    #pragma unroll
    for (int rr = 0; rr < RNUM; ++rr)
        tr[o][rr] = trains[((size_t)b * 256 + o) * RNUM + rr];
    __syncthreads();
    float acc = b5[o];
    for (int i = 0; i < 256; ++i) {
        #pragma unroll
        for (int rr = 0; rr < RNUM; ++rr)
            acc += w5[(o * 256 + i) * RNUM + rr] * tr[i][rr];
    }
    station[b * 256 + o] = acc;
}

// ---------------------------------------------------------------------------
// Broadcast sp_trains (ch 512..767) and sp_station (ch 768..1023)
// 8 consecutive floats per thread -> global_store_b128 pairs.
// ---------------------------------------------------------------------------
__global__ __launch_bounds__(256)
void bcast_kernel(const float* __restrict__ trains, const float* __restrict__ station,
                  float* __restrict__ out)
{
    const int c = blockIdx.x;   // 0..511
    const int r = blockIdx.y, b = blockIdx.z;
    const float v = (c < 256) ? trains[((size_t)b * 256 + c) * RNUM + r]
                              : station[b * 256 + (c - 256)];
    float4* orow4 = (float4*)(out + (((size_t)b * CHOUT + 512 + c) * RNUM + r) * NPTS);
    const int t = threadIdx.x;
    const float4 f4 = { v, v, v, v };
    orow4[t * 2]     = f4;
    orow4[t * 2 + 1] = f4;
}

// ---------------------------------------------------------------------------
// pws_g: gathered onehot(id_activa) (ch 1024..1039) and xyz (ch 1040..1042)
// idx read as int4 pairs, output as float4 pairs (consecutive per thread).
// ---------------------------------------------------------------------------
__global__ __launch_bounds__(256)
void pwsg_kernel(const float* __restrict__ x, const int* __restrict__ ida,
                 const int* __restrict__ idx, float* __restrict__ out)
{
    const int ch = blockIdx.x, r = blockIdx.y, b = blockIdx.z;
    const int4* irow4 = (const int4*)(idx + ((size_t)b * RNUM + r) * NPTS);
    float* orow = out + (((size_t)b * CHOUT + 1024 + ch) * RNUM + r) * NPTS;
    const int t = threadIdx.x;
    const int4 j0 = irow4[t * 2];
    const int4 j1 = irow4[t * 2 + 1];
    int js[8] = { j0.x, j0.y, j0.z, j0.w, j1.x, j1.y, j1.z, j1.w };
    float vv[8];
    #pragma unroll
    for (int u = 0; u < 8; ++u) {
        const int j = js[u];
        if (ch < RNUM) vv[u] = (ida[b * NPTS + j] == ch) ? 1.f : 0.f;
        else           vv[u] = x[((size_t)b * 3 + (ch - RNUM)) * NPTS + j];
    }
    float4* orow4 = (float4*)orow;
    float4 p0 = { vv[0], vv[1], vv[2], vv[3] };
    float4 p1 = { vv[4], vv[5], vv[6], vv[7] };
    orow4[t * 2]     = p0;
    orow4[t * 2 + 1] = p1;
}

// ---------------------------------------------------------------------------
extern "C" void kernel_launch(void* const* d_in, const int* in_sizes, int n_in,
                              void* d_out, int out_size, void* d_ws, size_t ws_size,
                              hipStream_t stream)
{
    (void)in_sizes; (void)n_in; (void)out_size; (void)ws_size;
    // setup_inputs() order: x, then params in insertion order (STN params unused)
    const float* x     = (const float*)d_in[0];
    const float* c1_w  = (const float*)d_in[1];
    const float* c1_b  = (const float*)d_in[2];
    const float* bn1_g = (const float*)d_in[3];
    const float* bn1_b = (const float*)d_in[4];
    const float* bn1_m = (const float*)d_in[5];
    const float* bn1_v = (const float*)d_in[6];
    const float* c2_w  = (const float*)d_in[7];
    const float* c2_b  = (const float*)d_in[8];
    const float* bn2_g = (const float*)d_in[9];
    const float* bn2_b = (const float*)d_in[10];
    const float* bn2_m = (const float*)d_in[11];
    const float* bn2_v = (const float*)d_in[12];
    const float* c3_w  = (const float*)d_in[13];
    const float* c3_b  = (const float*)d_in[14];
    const float* bn3_g = (const float*)d_in[15];
    const float* bn3_b = (const float*)d_in[16];
    const float* bn3_m = (const float*)d_in[17];
    const float* bn3_v = (const float*)d_in[18];
    const float* srt_w = (const float*)d_in[19];
    const float* srt_b = (const float*)d_in[20];
    const float* sp1_w = (const float*)d_in[21];
    const float* sp1_b = (const float*)d_in[22];
    const float* sp2_w = (const float*)d_in[23];
    const float* sp2_b = (const float*)d_in[24];
    const float* sp3_w = (const float*)d_in[25];
    const float* sp3_b = (const float*)d_in[26];
    const float* sp5_w = (const float*)d_in[27];
    const float* sp5_b = (const float*)d_in[28];

    // workspace layout (floats): ~8.3 MB total
    float* ws      = (float*)d_ws;
    float* h1      = ws;                                   // B*64*N
    float* h2      = h1 + (size_t)BATCH * 64 * NPTS;       // B*128*N
    float* h3      = h2 + (size_t)BATCH * 128 * NPTS;      // B*256*N
    float* val     = h3 + (size_t)BATCH * 256 * NPTS;      // B*R*N
    int*   idx     = (int*)(val + (size_t)BATCH * RNUM * NPTS); // B*R*N
    int*   ida     = idx + (size_t)BATCH * RNUM * NPTS;    // B*N
    float* cabins  = (float*)(ida + (size_t)BATCH * NPTS); // B*256*R*8
    float* trains  = cabins + (size_t)BATCH * 256 * RNUM * NCAB; // B*256*R
    float* station = trains + (size_t)BATCH * 256 * RNUM;  // B*256
    float* out     = (float*)d_out;

    layer1_kernel<<<BATCH * 64 * NPTS / 256, 256, 0, stream>>>(
        x, c1_w, c1_b, bn1_g, bn1_b, bn1_m, bn1_v, h1);

    dim3 g2(NPTS / 16, 128 / 16, BATCH);
    wmma_pointwise_conv<64, 128, true, true><<<g2, 32, 0, stream>>>(
        h1, c2_w, c2_b, bn2_g, bn2_b, bn2_m, bn2_v, h2, NPTS);

    dim3 g3(NPTS / 16, 256 / 16, BATCH);
    wmma_pointwise_conv<128, 256, true, false><<<g3, 32, 0, stream>>>(
        h2, c3_w, c3_b, bn3_g, bn3_b, bn3_m, bn3_v, h3, NPTS);

    dim3 g4(NPTS / 16, 1, BATCH);
    wmma_pointwise_conv<256, 16, false, false><<<g4, 32, 0, stream>>>(
        h3, srt_w, srt_b, nullptr, nullptr, nullptr, nullptr, val, NPTS);

    argmax_kernel<<<BATCH * NPTS / 256, 256, 0, stream>>>(val, ida);

    sort_kernel<<<dim3(RNUM, BATCH), 1024, 0, stream>>>(val, idx);

    gather_kernel<<<dim3(256, RNUM, BATCH), 256, 0, stream>>>(h3, idx, cabins, out);

    trains_kernel<<<dim3(RNUM, BATCH), 256, 0, stream>>>(
        cabins, sp1_w, sp1_b, sp2_w, sp2_b, sp3_w, sp3_b, trains);

    station_kernel<<<BATCH, 256, 0, stream>>>(trains, sp5_w, sp5_b, station);

    bcast_kernel<<<dim3(512, RNUM, BATCH), 256, 0, stream>>>(trains, station, out);

    pwsg_kernel<<<dim3(19, RNUM, BATCH), 256, 0, stream>>>(x, ida, idx, out);
}
